// AFT_FULL_28750511079849
// MI455X (gfx1250) — matrix operand order repaired and checked by reading, not compiled, not run on hardware
//
#include <hip/hip_runtime.h>
#include <hip/hip_bf16.h>
#include <math.h>

typedef _Float16 half_t;
typedef __attribute__((ext_vector_type(16))) _Float16 v16h;
typedef __attribute__((ext_vector_type(8)))  _Float16 v8h;
typedef __attribute__((ext_vector_type(8)))  float    v8f;
typedef __attribute__((ext_vector_type(4)))  int      v4i;

#define DEV __device__ __forceinline__

constexpr int BSZ  = 8;
constexpr int NSEQ = 2048;
constexpr int DM   = 512;

// ---- workspace layout (offsets in _Float16 elements) ----
constexpr size_t SZ_X    = (size_t)BSZ * NSEQ * DM;
constexpr size_t OFF_X1H = 0;
constexpr size_t OFF_X2H = OFF_X1H + SZ_X;
constexpr size_t OFF_WQ  = OFF_X2H + SZ_X;
constexpr size_t OFF_WK  = OFF_WQ  + (size_t)DM * DM;
constexpr size_t OFF_WV  = OFF_WK  + (size_t)DM * DM;
constexpr size_t OFF_EB  = OFF_WV  + (size_t)DM * DM;     // exp(position_biases) [t][s]
constexpr size_t OFF_SQT = OFF_EB  + (size_t)NSEQ * NSEQ; // sigmoid(q) transposed [b][d][t]
constexpr size_t OFF_EKT = OFF_SQT + SZ_X;                // exp(k)    transposed [b][d][s]
constexpr size_t OFF_EKVT= OFF_EKT + SZ_X;                // exp(k)*v  transposed [b][d][s]

// ---- stage-2 block-GEMM tiling ----
constexpr int T_BLK  = 128;            // t rows per block (16 per wave x 8 waves)
constexpr int D_BLK  = 64;             // d cols per block
constexpr int K_STEP = 32;
constexpr int NIT    = NSEQ / K_STEP;  // 64
constexpr int APAD   = 40;             // 32 + 8 halfs pad -> bank-conflict-free (stride 20 dwords)
constexpr int BPAD   = 40;
constexpr int ABUF   = T_BLK * APAD;         // 5120 halfs
constexpr int BUFSZ  = ABUF + 2 * D_BLK * BPAD; // 10240 halfs = 20KB; x2 buffers = 40KB LDS

// A-fragment: 16x32 f16 tile, row-major, row stride ld (halfs).
// ISA: lane L -> M=L&15; VGPR0..3 = K=(L>>4)*8..+7 ; VGPR4..7 = +16..+23
DEV v16h load_a16x32(const half_t* __restrict__ p0, int ld, int lane) {
  const int m  = lane & 15;
  const int kh = (lane >> 4) * 8;
  const half_t* p = p0 + (size_t)m * ld + kh;
  v8h lo = *(const v8h*)(p);
  v8h hi = *(const v8h*)(p + 16);
  v16h a;
#pragma unroll
  for (int i = 0; i < 8; ++i) { a[i] = lo[i]; a[i + 8] = hi[i]; }
  return a;
}

// B-fragment from B-transposed bt[n][k], row stride ldb.
// ISA: lane L -> N=L&15; elems 0..7 = K=(L>>4)*8..+7 ; elems 8..15 = +16..+23
DEV v16h load_bt32x16(const half_t* __restrict__ bt, int ldb, int k0, int lane) {
  const int n = lane & 15;
  const half_t* p = bt + (size_t)n * ldb + k0 + ((lane >> 4) << 3);
  v8h lo = *(const v8h*)(p);
  v8h hi = *(const v8h*)(p + 16);
  v16h b;
#pragma unroll
  for (int i = 0; i < 8; ++i) { b[i] = lo[i]; b[i + 8] = hi[i]; }
  return b;
}

DEV v8f wmma_f16(v16h a, v16h b, v8f c) {
  return __builtin_amdgcn_wmma_f32_16x16x32_f16(false, a, false, b, (short)0, c,
                                                false, false);
}

// ---- CDNA5 async global->LDS path (guarded; fallback = load + ds_store) ----
#if __has_builtin(__builtin_amdgcn_global_load_async_to_lds_b128)
#define USE_ASYNC_LDS 1
typedef __attribute__((address_space(1))) v4i* gv4i_p;  // clang: "__device__" AS1
typedef __attribute__((address_space(3))) v4i* sv4i_p;  // LDS AS3
DEV void async_copy_b128(const half_t* g, half_t* s) {
  __builtin_amdgcn_global_load_async_to_lds_b128((gv4i_p)(void*)g,
                                                 (sv4i_p)(void*)s, 0, 0);
}
template <int N> DEV void wait_asynccnt() {
#if __has_builtin(__builtin_amdgcn_s_wait_asynccnt)
  __builtin_amdgcn_s_wait_asynccnt(N);
#else
  asm volatile("s_wait_asynccnt %0" ::"i"(N) : "memory");
#endif
}
#else
#define USE_ASYNC_LDS 0
#endif

// ---------------- Stage 0: convert / exponentiate ----------------
__global__ void aft_prep_kernel(const float* __restrict__ x1,
                                const float* __restrict__ x2,
                                const float* __restrict__ wq,
                                const float* __restrict__ wk,
                                const float* __restrict__ wv,
                                const float* __restrict__ pb,
                                half_t* __restrict__ ws) {
  const size_t i0     = (size_t)blockIdx.x * blockDim.x + threadIdx.x;
  const size_t stride = (size_t)gridDim.x * blockDim.x;
  for (size_t t = i0; t < SZ_X; t += stride) {
    ws[OFF_X1H + t] = (half_t)x1[t];
    ws[OFF_X2H + t] = (half_t)x2[t];
  }
  for (size_t t = i0; t < (size_t)DM * DM; t += stride) {
    ws[OFF_WQ + t] = (half_t)wq[t];
    ws[OFF_WK + t] = (half_t)wk[t];
    ws[OFF_WV + t] = (half_t)wv[t];
  }
  for (size_t t = i0; t < (size_t)NSEQ * NSEQ; t += stride) {
    ws[OFF_EB + t] = (half_t)__expf(pb[t]);
  }
}

// ---------------- Stage 1: QKV linears + pointwise, transposed f16 out ----------------
// One wave: 16(seq) x 16(dmodel) tile of q,k,v simultaneously (48 WMMAs).
__global__ void aft_qkv_kernel(const float* __restrict__ bq,
                               const float* __restrict__ bk,
                               const float* __restrict__ bv,
                               half_t* __restrict__ ws) {
  const int lane = threadIdx.x & 31;
  const int wave = blockIdx.x * (blockDim.x >> 5) + (threadIdx.x >> 5);
  const int jt = wave & 31;
  const int nt = (wave >> 5) & 127;
  const int b  = wave >> 12;
  if (b >= BSZ) return;  // wave-uniform
  const int n0 = nt * 16, j0 = jt * 16;

  const half_t* x1  = ws + OFF_X1H + ((size_t)b * NSEQ + n0) * DM;
  const half_t* x2  = ws + OFF_X2H + ((size_t)b * NSEQ + n0) * DM;
  const half_t* wqh = ws + OFF_WQ + (size_t)j0 * DM;
  const half_t* wkh = ws + OFF_WK + (size_t)j0 * DM;
  const half_t* wvh = ws + OFF_WV + (size_t)j0 * DM;

  v8f cq = {}, ck = {}, cv = {};
#pragma unroll 4
  for (int k0 = 0; k0 < DM; k0 += 32) {
    v16h a1 = load_a16x32(x1 + k0, DM, lane);
    v16h a2 = load_a16x32(x2 + k0, DM, lane);
    v16h fq = load_bt32x16(wqh, DM, k0, lane);
    v16h fk = load_bt32x16(wkh, DM, k0, lane);
    v16h fv = load_bt32x16(wvh, DM, k0, lane);
    cq = wmma_f16(a1, fq, cq);
    ck = wmma_f16(a1, fk, ck);
    cv = wmma_f16(a2, fv, cv);
  }

  const int jcol  = j0 + (lane & 15);
  const int nbase = n0 + ((lane >> 4) << 3);
  const float biasq = bq[jcol], biask = bk[jcol], biasv = bv[jcol];
  v8h o_sq, o_ek, o_ekv;
#pragma unroll
  for (int i = 0; i < 8; ++i) {
    float q  = cq[i] + biasq;
    float e  = __expf(ck[i] + biask);
    float vv = cv[i] + biasv;
    o_sq[i]  = (half_t)(1.0f / (1.0f + __expf(-q)));
    o_ek[i]  = (half_t)e;
    o_ekv[i] = (half_t)(e * vv);
  }
  const size_t rowoff = ((size_t)b * DM + jcol) * NSEQ + nbase;
  *(v8h*)(ws + OFF_SQT  + rowoff) = o_sq;
  *(v8h*)(ws + OFF_EKT  + rowoff) = o_ek;
  *(v8h*)(ws + OFF_EKVT + rowoff) = o_ekv;
}

// ---------------- Stage 2: LDS-staged block GEMM ----------------
// Block tile: 128(t) x 64(d); wave w owns rows t0+16w..+15, all 64 d.
// Per k-step stage A(128x32) + Bnum(64x32) + Bden(64x32) into LDS once,
// double-buffered, one barrier per step; async loads overlap compute.
__global__ void aft_out_kernel(float* __restrict__ out,
                               const half_t* __restrict__ ws) {
  __shared__ half_t smem[2 * BUFSZ];

  const int tid  = threadIdx.x;
  const int lane = tid & 31;
  const int w    = tid >> 5;
  const int bid  = blockIdx.x;
  const int dt = bid & 7, tb = (bid >> 3) & 15, b = bid >> 7;
  const int t0 = tb * T_BLK, d0 = dt * D_BLK;

  const half_t* eb_base   = ws + OFF_EB   + (size_t)t0 * NSEQ;
  const half_t* ekvt_base = ws + OFF_EKVT + ((size_t)b * DM + d0) * NSEQ;
  const half_t* ekt_base  = ws + OFF_EKT  + ((size_t)b * DM + d0) * NSEQ;

  // Staging role: threads 0..127 each own one A row (64B); threads 128..255
  // own one B row (rows 0..63 = num panel, 64..127 = den panel). Wave-uniform split.
  const int  r    = tid & 127;
  const half_t* grow =
      (tid < 128) ? (eb_base + (size_t)r * NSEQ)
                  : ((r < 64) ? (ekvt_base + (size_t)r * NSEQ)
                              : (ekt_base + (size_t)(r - 64) * NSEQ));
  const int loff = (tid < 128) ? (r * APAD) : (ABUF + r * BPAD);

  v8f num[4] = {}, den[4] = {};

  const int kh = (lane >> 4) * 8;                       // A k-offset
  const int aRow = (w * 16 + (lane & 15)) * APAD;       // A row in LDS
  const int bRow = (lane & 15) * BPAD + ((lane >> 4) << 3); // B base in LDS

#if USE_ASYNC_LDS
  // ---- async global->LDS pipeline (ASYNCcnt + s_wait_asynccnt) ----
  {
    half_t* dst = smem + loff;
#pragma unroll
    for (int q = 0; q < 4; ++q) async_copy_b128(grow + q * 8, dst + q * 8);
  }
  for (int it = 0; it < NIT; ++it) {
    const int cur = (it & 1) * BUFSZ;
    const int nxt = ((it + 1) & 1) * BUFSZ;
    wait_asynccnt<0>();
    __syncthreads();  // buf[cur] complete everywhere; buf[nxt] reads all retired
    if (it + 1 < NIT) {
      const half_t* g = grow + (it + 1) * K_STEP;
      half_t* dst = smem + nxt + loff;
#pragma unroll
      for (int q = 0; q < 4; ++q) async_copy_b128(g + q * 8, dst + q * 8);
    }
#else
  // ---- fallback: global->VGPR->ds_store pipeline ----
  v8h s0r, s1r, s2r, s3r;
  {
    const half_t* g = grow;
    s0r = *(const v8h*)(g);      s1r = *(const v8h*)(g + 8);
    s2r = *(const v8h*)(g + 16); s3r = *(const v8h*)(g + 24);
  }
  for (int it = 0; it < NIT; ++it) {
    const int cur = (it & 1) * BUFSZ;
    {
      half_t* p = smem + cur + loff;
      *(v8h*)(p) = s0r;      *(v8h*)(p + 8) = s1r;
      *(v8h*)(p + 16) = s2r; *(v8h*)(p + 24) = s3r;
    }
    if (it + 1 < NIT) {
      const half_t* g = grow + (it + 1) * K_STEP;
      s0r = *(const v8h*)(g);      s1r = *(const v8h*)(g + 8);
      s2r = *(const v8h*)(g + 16); s3r = *(const v8h*)(g + 24);
    }
    __syncthreads();  // buf[cur] visible; also separates buffer reuse
#endif
    // ---- compute: 1 A-fragment (reused 8x) + 8 B-fragments -> 8 WMMAs ----
    {
      const half_t* sA = smem + cur + aRow;
      v8h alo = *(const v8h*)(sA + kh);
      v8h ahi = *(const v8h*)(sA + kh + 16);
      v16h a;
#pragma unroll
      for (int i = 0; i < 8; ++i) { a[i] = alo[i]; a[i + 8] = ahi[i]; }

      const half_t* sB = smem + cur + ABUF + bRow;
#pragma unroll
      for (int c = 0; c < 4; ++c) {
        const half_t* pn = sB + c * 16 * BPAD;
        const half_t* pd = pn + 64 * BPAD;
        v8h nlo = *(const v8h*)(pn), nhi = *(const v8h*)(pn + 16);
        v8h dlo = *(const v8h*)(pd), dhi = *(const v8h*)(pd + 16);
        v16h bn, bd;
#pragma unroll
        for (int i = 0; i < 8; ++i) {
          bn[i] = nlo[i]; bn[i + 8] = nhi[i];
          bd[i] = dlo[i]; bd[i + 8] = dhi[i];
        }
        num[c] = wmma_f16(a, bn, num[c]);
        den[c] = wmma_f16(a, bd, den[c]);
      }
    }
  }

  // ---- epilogue: out = sigmoid(q) * num / den ----
  const int tbase = t0 + w * 16 + ((lane >> 4) << 3);
#pragma unroll
  for (int c = 0; c < 4; ++c) {
    const int d = d0 + c * 16 + (lane & 15);
    v8h sq = *(const v8h*)(ws + OFF_SQT + ((size_t)b * DM + d) * NSEQ + tbase);
#pragma unroll
    for (int i = 0; i < 8; ++i) {
      out[((size_t)b * NSEQ + tbase + i) * DM + d] =
          (float)sq[i] * (num[c][i] / den[c][i]);
    }
  }
}

extern "C" void kernel_launch(void* const* d_in, const int* in_sizes, int n_in,
                              void* d_out, int out_size, void* d_ws, size_t ws_size,
                              hipStream_t stream) {
  const float* x1 = (const float*)d_in[0];
  const float* x2 = (const float*)d_in[1];
  const float* Wq = (const float*)d_in[2];
  const float* bq = (const float*)d_in[3];
  const float* Wk = (const float*)d_in[4];
  const float* bk = (const float*)d_in[5];
  const float* Wv = (const float*)d_in[6];
  const float* bv = (const float*)d_in[7];
  const float* pb = (const float*)d_in[8];
  float*  out = (float*)d_out;
  half_t* ws  = (half_t*)d_ws;

  aft_prep_kernel<<<2048, 256, 0, stream>>>(x1, x2, Wq, Wk, Wv, pb, ws);
  aft_qkv_kernel<<<4096, 256, 0, stream>>>(bq, bk, bv, ws);   // 32768 waves
  // 8 batches x 16 t-blocks x 8 d-blocks = 1024 blocks (8 waves each)
  aft_out_kernel<<<1024, 256, 0, stream>>>(out, ws);
}
